// MLayer_70635032150161
// MI455X (gfx1250) — matrix operand order repaired and checked by reading, not compiled, not run on hardware
//
#include <hip/hip_runtime.h>

// ---------------------------------------------------------------------------
// Geodesic Heun integrator + geo GEMMs for MI455X (gfx1250, wave32).
//  Kernel 1: fused LN + gate MLP + low-rank Christoffel + Heun
//            -> staged x_next / v_next as PRE-SPLIT bf16 hi/lo planes in d_ws
//  Kernel W: one-shot split of wx / wv into bf16 hi/lo planes
//  Kernel 2: out = state @ W^T + b, bf16 WMMA 3-term split product,
//            double-buffered LDS fed by GLOBAL_LOAD_ASYNC_TO_LDS_B128.
// Assumes ws_size >= 4*B*D*2 + 4*D*D*2 bytes (264 MB).
// ---------------------------------------------------------------------------

#define BATCH 32768
#define DIM   1024
#define NHEAD 4
#define HD    256
#define HD4   64
#define RNK   4
#define EPSV  1e-5f

typedef __attribute__((ext_vector_type(16))) __bf16 v16bf;
typedef __attribute__((ext_vector_type(8)))  float  v8f;

union BfPack {
  uint4 q[2];
  v16bf v;
};

// split f32 -> bf16(hi) + bf16(lo), truncation both times (err ~2^-17 rel)
__device__ __forceinline__ void split_pack4(float a, float b, float c, float d,
                                            uint2& ph, uint2& pl)
{
  const unsigned int u0 = __float_as_uint(a), u1 = __float_as_uint(b);
  const unsigned int u2 = __float_as_uint(c), u3 = __float_as_uint(d);
  const float l0f = a - __uint_as_float(u0 & 0xFFFF0000u);
  const float l1f = b - __uint_as_float(u1 & 0xFFFF0000u);
  const float l2f = c - __uint_as_float(u2 & 0xFFFF0000u);
  const float l3f = d - __uint_as_float(u3 & 0xFFFF0000u);
  ph.x = ((u1 >> 16) << 16) | (u0 >> 16);
  ph.y = ((u3 >> 16) << 16) | (u2 >> 16);
  pl.x = ((__float_as_uint(l1f) >> 16) << 16) | (__float_as_uint(l0f) >> 16);
  pl.y = ((__float_as_uint(l3f) >> 16) << 16) | (__float_as_uint(l2f) >> 16);
}

// ===========================================================================
// Kernel 1: fused state update. One block (256 thr, 8 waves) per 4 batch rows.
// Thread t owns elements d=4t..4t+3 of every row; head h = t/64.
// ===========================================================================
__global__ __launch_bounds__(256) void fused_heun_kernel(
    const float* __restrict__ xg, const float* __restrict__ vg,
    const float* __restrict__ fg,
    const float* __restrict__ lnxg, const float* __restrict__ lnxb,
    const float* __restrict__ lnvg, const float* __restrict__ lnvb,
    const float* __restrict__ gw1, const float* __restrict__ gb1,
    const float* __restrict__ gw2, const float* __restrict__ gb2,
    const float* __restrict__ cu,  const float* __restrict__ cv,
    const float* __restrict__ cw,  const float* __restrict__ dtp,
    unsigned short* __restrict__ wsxh, unsigned short* __restrict__ wsxl,
    unsigned short* __restrict__ wsvh, unsigned short* __restrict__ wsvl)
{
  __shared__ float  s_x[4][DIM];
  __shared__ float4 s_ru[256];
  __shared__ float4 s_rw[256];
  __shared__ float  s_h1[4][NHEAD][HD4];
  __shared__ float  s_dt[4][NHEAD];
  __shared__ float4 s_uw[4][NHEAD];

  const int t  = threadIdx.x;
  const int d  = t << 2;
  const int h  = t >> 6;
  const int dm = d & (HD - 1);
  const size_t row0 = (size_t)blockIdx.x * 4;

  float vh[4][4], fh[4][4];
  float mx[4], rxs[4], mv[4], rvs[4];

  // ---- load rows + LayerNorm statistics (block-wide tree per row) ----
  #pragma unroll
  for (int r = 0; r < 4; ++r) {
    const float4 x4 = *(const float4*)(xg + (row0 + r) * DIM + d);
    const float4 v4 = *(const float4*)(vg + (row0 + r) * DIM + d);
    const float4 f4 = *(const float4*)(fg + (row0 + r) * DIM + d);
    s_x[r][d+0]=x4.x; s_x[r][d+1]=x4.y; s_x[r][d+2]=x4.z; s_x[r][d+3]=x4.w;
    vh[r][0]=v4.x; vh[r][1]=v4.y; vh[r][2]=v4.z; vh[r][3]=v4.w;
    fh[r][0]=f4.x; fh[r][1]=f4.y; fh[r][2]=f4.z; fh[r][3]=f4.w;
    float4 p;
    p.x = x4.x + x4.y + x4.z + x4.w;
    p.y = x4.x*x4.x + x4.y*x4.y + x4.z*x4.z + x4.w*x4.w;
    p.z = v4.x + v4.y + v4.z + v4.w;
    p.w = v4.x*v4.x + v4.y*v4.y + v4.z*v4.z + v4.w*v4.w;
    s_ru[t] = p;
    __syncthreads();
    for (int s = 128; s > 0; s >>= 1) {
      if (t < s) {
        float4 a = s_ru[t]; const float4 b = s_ru[t + s];
        a.x += b.x; a.y += b.y; a.z += b.z; a.w += b.w;
        s_ru[t] = a;
      }
      __syncthreads();
    }
    const float4 tot = s_ru[0];
    __syncthreads();
    mx[r]  = tot.x * (1.0f / DIM);
    rxs[r] = rsqrtf(tot.y * (1.0f / DIM) - mx[r] * mx[r] + EPSV);
    mv[r]  = tot.z * (1.0f / DIM);
    rvs[r] = rsqrtf(tot.w * (1.0f / DIM) - mv[r] * mv[r] + EPSV);
  }

  // ---- normalize ----
  const float4 g_x = *(const float4*)(lnxg + d);
  const float4 b_x = *(const float4*)(lnxb + d);
  const float4 g_v = *(const float4*)(lnvg + d);
  const float4 b_v = *(const float4*)(lnvb + d);
  #pragma unroll
  for (int r = 0; r < 4; ++r) {
    s_x[r][d+0] = (s_x[r][d+0] - mx[r]) * rxs[r] * g_x.x + b_x.x;
    s_x[r][d+1] = (s_x[r][d+1] - mx[r]) * rxs[r] * g_x.y + b_x.y;
    s_x[r][d+2] = (s_x[r][d+2] - mx[r]) * rxs[r] * g_x.z + b_x.z;
    s_x[r][d+3] = (s_x[r][d+3] - mx[r]) * rxs[r] * g_x.w + b_x.w;
    vh[r][0] = (vh[r][0] - mv[r]) * rvs[r] * g_v.x + b_v.x;
    vh[r][1] = (vh[r][1] - mv[r]) * rvs[r] * g_v.y + b_v.y;
    vh[r][2] = (vh[r][2] - mv[r]) * rvs[r] * g_v.z + b_v.z;
    vh[r][3] = (vh[r][3] - mv[r]) * rvs[r] * g_v.w + b_v.w;
  }
  __syncthreads();

  // ---- gating MLP: thread t = hidden unit (h, e) for all 4 rows ----
  {
    const int e = t & 63;
    const float bb = gb1[h * HD4 + e];
    float acc[4];
    #pragma unroll
    for (int r = 0; r < 4; ++r) acc[r] = bb;
    const float* wp = gw1 + (size_t)h * HD * HD4 + e;
    for (int dd = 0; dd < HD; ++dd) {
      const float w = wp[dd * HD4];
      #pragma unroll
      for (int r = 0; r < 4; ++r) acc[r] += s_x[r][h * HD + dd] * w;
    }
    #pragma unroll
    for (int r = 0; r < 4; ++r) s_h1[r][h][e] = tanhf(acc[r]);
  }
  __syncthreads();
  if (t < 16) {
    const int r = t >> 2, hh = t & 3;
    float zacc = gb2[hh];
    for (int e = 0; e < HD4; ++e) zacc += s_h1[r][hh][e] * gw2[hh * HD4 + e];
    const float gate = 1.0f / (1.0f + expf(-zacc));
    const float sp   = log1pf(expf(dtp[hh]));   // softplus
    s_dt[r][hh] = sp * gate;
  }
  __syncthreads();
  float dtr[4];
  #pragma unroll
  for (int r = 0; r < 4; ++r) dtr[r] = s_dt[r][h];

  float k1[4][4], ve[4][4];

  // ---- Gamma(vh) ----
  {
    float pu[4][4], pw[4][4];
    #pragma unroll
    for (int r = 0; r < 4; ++r)
      #pragma unroll
      for (int rr = 0; rr < 4; ++rr) { pu[r][rr] = 0.f; pw[r][rr] = 0.f; }
    #pragma unroll
    for (int rr = 0; rr < RNK; ++rr)
      #pragma unroll
      for (int j = 0; j < 4; ++j) {
        const float cuv = cu[(size_t)(h * RNK + rr) * HD + dm + j];
        const float cvv = cv[(size_t)(h * RNK + rr) * HD + dm + j];
        #pragma unroll
        for (int r = 0; r < 4; ++r) {
          pu[r][rr] += vh[r][j] * cuv;
          pw[r][rr] += vh[r][j] * cvv;
        }
      }
    for (int r = 0; r < 4; ++r) {
      s_ru[t] = make_float4(pu[r][0], pu[r][1], pu[r][2], pu[r][3]);
      s_rw[t] = make_float4(pw[r][0], pw[r][1], pw[r][2], pw[r][3]);
      __syncthreads();
      for (int s = 32; s > 0; s >>= 1) {
        if ((t & 63) < s) {
          float4 a = s_ru[t]; const float4 b = s_ru[t + s];
          a.x += b.x; a.y += b.y; a.z += b.z; a.w += b.w; s_ru[t] = a;
          float4 c = s_rw[t]; const float4 e2 = s_rw[t + s];
          c.x += e2.x; c.y += e2.y; c.z += e2.z; c.w += e2.w; s_rw[t] = c;
        }
        __syncthreads();
      }
      if ((t & 63) == 0) {
        const float4 a = s_ru[t], b = s_rw[t];
        s_uw[r][t >> 6] = make_float4(a.x*b.x, a.y*b.y, a.z*b.z, a.w*b.w);
      }
      __syncthreads();
    }
    #pragma unroll
    for (int r = 0; r < 4; ++r) {
      const float4 uw = s_uw[r][h];
      #pragma unroll
      for (int j = 0; j < 4; ++j) {
        const float4 cw4 = *(const float4*)(cw + (size_t)(h * HD + dm + j) * RNK);
        const float g = uw.x*cw4.x + uw.y*cw4.y + uw.z*cw4.z + uw.w*cw4.w;
        k1[r][j] = fh[r][j] - g;
        ve[r][j] = vh[r][j] + dtr[r] * k1[r][j];
      }
    }
  }
  __syncthreads();

  // ---- Gamma(ve), k2, Heun outputs (pre-split bf16 hi/lo) ----
  {
    float pu[4][4], pw[4][4];
    #pragma unroll
    for (int r = 0; r < 4; ++r)
      #pragma unroll
      for (int rr = 0; rr < 4; ++rr) { pu[r][rr] = 0.f; pw[r][rr] = 0.f; }
    #pragma unroll
    for (int rr = 0; rr < RNK; ++rr)
      #pragma unroll
      for (int j = 0; j < 4; ++j) {
        const float cuv = cu[(size_t)(h * RNK + rr) * HD + dm + j];
        const float cvv = cv[(size_t)(h * RNK + rr) * HD + dm + j];
        #pragma unroll
        for (int r = 0; r < 4; ++r) {
          pu[r][rr] += ve[r][j] * cuv;
          pw[r][rr] += ve[r][j] * cvv;
        }
      }
    for (int r = 0; r < 4; ++r) {
      s_ru[t] = make_float4(pu[r][0], pu[r][1], pu[r][2], pu[r][3]);
      s_rw[t] = make_float4(pw[r][0], pw[r][1], pw[r][2], pw[r][3]);
      __syncthreads();
      for (int s = 32; s > 0; s >>= 1) {
        if ((t & 63) < s) {
          float4 a = s_ru[t]; const float4 b = s_ru[t + s];
          a.x += b.x; a.y += b.y; a.z += b.z; a.w += b.w; s_ru[t] = a;
          float4 c = s_rw[t]; const float4 e2 = s_rw[t + s];
          c.x += e2.x; c.y += e2.y; c.z += e2.z; c.w += e2.w; s_rw[t] = c;
        }
        __syncthreads();
      }
      if ((t & 63) == 0) {
        const float4 a = s_ru[t], b = s_rw[t];
        s_uw[r][t >> 6] = make_float4(a.x*b.x, a.y*b.y, a.z*b.z, a.w*b.w);
      }
      __syncthreads();
    }
    #pragma unroll
    for (int r = 0; r < 4; ++r) {
      const float4 uw = s_uw[r][h];
      float xn[4], vn[4];
      #pragma unroll
      for (int j = 0; j < 4; ++j) {
        const float4 cw4 = *(const float4*)(cw + (size_t)(h * HD + dm + j) * RNK);
        const float g  = uw.x*cw4.x + uw.y*cw4.y + uw.z*cw4.z + uw.w*cw4.w;
        const float k2 = fh[r][j] - g;
        vn[j] = vh[r][j] + dtr[r] * 0.5f * (k1[r][j] + k2);
        xn[j] = s_x[r][d + j] + dtr[r] * 0.5f * (vh[r][j] + ve[r][j]);
      }
      const size_t idx = (row0 + r) * DIM + d;
      uint2 ph, pl;
      split_pack4(xn[0], xn[1], xn[2], xn[3], ph, pl);
      *(uint2*)(wsxh + idx) = ph;
      *(uint2*)(wsxl + idx) = pl;
      split_pack4(vn[0], vn[1], vn[2], vn[3], ph, pl);
      *(uint2*)(wsvh + idx) = ph;
      *(uint2*)(wsvl + idx) = pl;
    }
  }
}

// ===========================================================================
// Kernel W: one-shot split of the two 1024x1024 weight matrices.
// ===========================================================================
__global__ __launch_bounds__(256) void split_weights_kernel(
    const float* __restrict__ wxm, const float* __restrict__ wvm,
    unsigned short* __restrict__ wxh, unsigned short* __restrict__ wxl,
    unsigned short* __restrict__ wvh, unsigned short* __restrict__ wvl)
{
  const size_t i = ((size_t)blockIdx.x * 256 + threadIdx.x) * 4;
  const float4 a = *(const float4*)(wxm + i);
  const float4 b = *(const float4*)(wvm + i);
  uint2 ph, pl;
  split_pack4(a.x, a.y, a.z, a.w, ph, pl);
  *(uint2*)(wxh + i) = ph;
  *(uint2*)(wxl + i) = pl;
  split_pack4(b.x, b.y, b.z, b.w, ph, pl);
  *(uint2*)(wvh + i) = ph;
  *(uint2*)(wvl + i) = pl;
}

// ===========================================================================
// Kernel 2: O[M,N] = S[M,K] @ W[N,K]^T + bias. Pre-split bf16 operands are
// DMA'd global->LDS with GLOBAL_LOAD_ASYNC_TO_LDS_B128 (ASYNCcnt), double-
// buffered; compute is 24 v_wmma_f32_16x16x32_bf16 per wave per K-step.
// ===========================================================================
#define BM 128
#define BN 128
#define BK 32
#define ROWB 80                       // padded LDS row: 40 ushorts (16B-aligned,
                                      // 20-bank stride -> conflict-free b128)
#define ARR_BYTES (BM * ROWB)         // 10240
#define OFF_AH 0
#define OFF_AL (1 * ARR_BYTES)
#define OFF_BH (2 * ARR_BYTES)
#define OFF_BL (3 * ARR_BYTES)
#define STAGE_BYTES (4 * ARR_BYTES)   // 40960

// Async copy 16B/lane global->LDS. The LDS operand is derived from the real
// shared-memory pointer: ptrtoint CAPTURES the LDS object (so the "memory"
// clobber forces alias analysis to assume the asm writes it -- without this
// the subsequent ds_loads fold to undef), and per the aperture scheme the low
// 32 bits of a flat shared address ARE the LDS byte offset the VDST expects.
__device__ __forceinline__ void async_cp16(const void* lds_ptr,
                                           const unsigned char* base,
                                           unsigned goff)
{
  const unsigned lds_off = (unsigned)(size_t)lds_ptr;
  asm volatile("global_load_async_to_lds_b128 %0, %1, %2"
               :: "v"(lds_off), "v"(goff), "s"(base)
               : "memory");
}

__device__ __forceinline__ void wait_async0()
{
#if __has_builtin(__builtin_amdgcn_s_wait_asynccnt)
  __builtin_amdgcn_s_wait_asynccnt(0);
#else
  asm volatile("s_wait_asynccnt 0x0" ::: "memory");
#endif
}

__global__ __launch_bounds__(256) void geo_gemm_kernel(
    const unsigned short* __restrict__ wsxh, const unsigned short* __restrict__ wsxl,
    const unsigned short* __restrict__ wsvh, const unsigned short* __restrict__ wsvl,
    const unsigned short* __restrict__ wxh,  const unsigned short* __restrict__ wxl,
    const unsigned short* __restrict__ wvh,  const unsigned short* __restrict__ wvl,
    const float* __restrict__ bxv, const float* __restrict__ bvv,
    float* __restrict__ out)
{
  __shared__ __align__(16) unsigned char smbuf[2 * STAGE_BYTES];

  const int zsel = blockIdx.z;
  const unsigned char* Sh = (const unsigned char*)(zsel ? wsvh : wsxh);
  const unsigned char* Sl = (const unsigned char*)(zsel ? wsvl : wsxl);
  const unsigned char* Wh = (const unsigned char*)(zsel ? wvh  : wxh);
  const unsigned char* Wl = (const unsigned char*)(zsel ? wvl  : wxl);
  const float* __restrict__ bias = zsel ? bvv : bxv;
  float* __restrict__ O = out + (size_t)zsel * BATCH * DIM;

  const int t    = threadIdx.x;
  const int lane = t & 31;
  const int wid  = t >> 5;
  const int bm0  = blockIdx.y * BM;
  const int bn0  = blockIdx.x * BN;

  v8f acc[8];
  #pragma unroll
  for (int s = 0; s < 8; ++s)
    #pragma unroll
    for (int i = 0; i < 8; ++i) acc[s][i] = 0.0f;

  // 512 16-byte chunks per 128x32 bf16 tile; thread t copies chunks t, t+256
  const int r0 = t >> 2,          j0 = t & 3;
  const int r1 = (t + 256) >> 2,  j1 = t & 3;
  const unsigned la0 = (unsigned)(r0 * ROWB + j0 * 16);
  const unsigned la1 = (unsigned)(r1 * ROWB + j1 * 16);
  const unsigned gA0 = (unsigned)(2 * ((bm0 + r0) * DIM) + j0 * 16);
  const unsigned gA1 = (unsigned)(2 * ((bm0 + r1) * DIM) + j1 * 16);
  const unsigned gB0 = (unsigned)(2 * ((bn0 + r0) * DIM) + j0 * 16);
  const unsigned gB1 = (unsigned)(2 * ((bn0 + r1) * DIM) + j1 * 16);

  // per-lane WMMA gather coordinates (documented CDNA5 wave32 layouts)
  const int am = wid * 16 + (lane & 15);
  const unsigned aoff = (unsigned)(am * ROWB + ((lane >> 4) ? 16 : 0));
  const unsigned bsel = (unsigned)((lane >> 4) ? 16 : 0);
  const unsigned nrow = (unsigned)((lane & 15) * ROWB);

  // prologue: stage 0 copies for k0 = 0
  async_cp16(smbuf + OFF_AH + la0, Sh, gA0);
  async_cp16(smbuf + OFF_AH + la1, Sh, gA1);
  async_cp16(smbuf + OFF_AL + la0, Sl, gA0);
  async_cp16(smbuf + OFF_AL + la1, Sl, gA1);
  async_cp16(smbuf + OFF_BH + la0, Wh, gB0);
  async_cp16(smbuf + OFF_BH + la1, Wh, gB1);
  async_cp16(smbuf + OFF_BL + la0, Wl, gB0);
  async_cp16(smbuf + OFF_BL + la1, Wl, gB1);

  unsigned stage = 0;
  for (int k0 = 0; k0 < DIM; k0 += BK) {
    wait_async0();       // my copies for this stage have landed in LDS
    __syncthreads();     // everyone's copies landed; prev compute on other
                         // stage finished -> safe to refill it
    if (k0 + BK < DIM) {
      const unsigned sb = (stage ^ 1u) * STAGE_BYTES;
      const unsigned kb = (unsigned)(2 * (k0 + BK));
      async_cp16(smbuf + sb + OFF_AH + la0, Sh + kb, gA0);
      async_cp16(smbuf + sb + OFF_AH + la1, Sh + kb, gA1);
      async_cp16(smbuf + sb + OFF_AL + la0, Sl + kb, gA0);
      async_cp16(smbuf + sb + OFF_AL + la1, Sl + kb, gA1);
      async_cp16(smbuf + sb + OFF_BH + la0, Wh + kb, gB0);
      async_cp16(smbuf + sb + OFF_BH + la1, Wh + kb, gB1);
      async_cp16(smbuf + sb + OFF_BL + la0, Wl + kb, gB0);
      async_cp16(smbuf + sb + OFF_BL + la1, Wl + kb, gB1);
    }

    const unsigned sb = stage * STAGE_BYTES;
    // A gather: lane L = row M=L%16; halves j<8 -> K=ac0+j, j>=8 -> 16+ac0+j-8
    BfPack ah, al;
    ah.q[0] = *(const uint4*)(smbuf + sb + OFF_AH + aoff);
    ah.q[1] = *(const uint4*)(smbuf + sb + OFF_AH + aoff + 32);
    al.q[0] = *(const uint4*)(smbuf + sb + OFF_AL + aoff);
    al.q[1] = *(const uint4*)(smbuf + sb + OFF_AL + aoff + 32);

    #pragma unroll
    for (int s = 0; s < 8; ++s) {
      // B gather: lane L = col N=L%16; halves j<8 -> K=bc0+j, j>=8 -> 16+bc0+j-8
      const unsigned boff = (unsigned)(s * 16 * ROWB) + nrow + bsel;
      BfPack bh, bl;
      bh.q[0] = *(const uint4*)(smbuf + sb + OFF_BH + boff);
      bh.q[1] = *(const uint4*)(smbuf + sb + OFF_BH + boff + 32);
      bl.q[0] = *(const uint4*)(smbuf + sb + OFF_BL + boff);
      bl.q[1] = *(const uint4*)(smbuf + sb + OFF_BL + boff + 32);
      acc[s] = __builtin_amdgcn_wmma_f32_16x16x32_bf16(
                   false, al.v, false, bh.v, (short)0, acc[s], false, false);
      acc[s] = __builtin_amdgcn_wmma_f32_16x16x32_bf16(
                   false, ah.v, false, bl.v, (short)0, acc[s], false, false);
      acc[s] = __builtin_amdgcn_wmma_f32_16x16x32_bf16(
                   false, ah.v, false, bh.v, (short)0, acc[s], false, false);
    }
    stage ^= 1u;
  }

  // C/D layout: VGPR i -> M = i + (lane<16 ? 0 : 8), N = lane%16
  const int n_in = lane & 15;
  const int mOff = (lane >> 4) << 3;
  #pragma unroll
  for (int s = 0; s < 8; ++s) {
    const int n = bn0 + s * 16 + n_in;
    const float bb = bias[n];
    #pragma unroll
    for (int i = 0; i < 8; ++i) {
      const int m = bm0 + wid * 16 + mOff + i;
      __builtin_nontemporal_store(acc[s][i] + bb, &O[(size_t)m * DIM + n]);
    }
  }
}

// ===========================================================================
extern "C" void kernel_launch(void* const* d_in, const int* in_sizes, int n_in,
                              void* d_out, int out_size, void* d_ws, size_t ws_size,
                              hipStream_t stream) {
  (void)in_sizes; (void)n_in; (void)out_size; (void)ws_size;
  const float* xg   = (const float*)d_in[0];
  const float* vg   = (const float*)d_in[1];
  const float* fg   = (const float*)d_in[2];
  const float* lnxg = (const float*)d_in[3];
  const float* lnxb = (const float*)d_in[4];
  const float* lnvg = (const float*)d_in[5];
  const float* lnvb = (const float*)d_in[6];
  const float* gw1  = (const float*)d_in[7];
  const float* gb1  = (const float*)d_in[8];
  const float* gw2  = (const float*)d_in[9];
  const float* gb2  = (const float*)d_in[10];
  const float* cu   = (const float*)d_in[11];
  const float* cv   = (const float*)d_in[12];
  const float* cw   = (const float*)d_in[13];
  const float* dtp  = (const float*)d_in[14];
  const float* wxm  = (const float*)d_in[15];
  const float* bxv  = (const float*)d_in[16];
  const float* wvm  = (const float*)d_in[17];
  const float* bvv  = (const float*)d_in[18];

  const size_t nsd = (size_t)BATCH * DIM;
  const size_t nww = (size_t)DIM * DIM;
  unsigned short* wsxh = (unsigned short*)d_ws;
  unsigned short* wsxl = wsxh + nsd;
  unsigned short* wsvh = wsxl + nsd;
  unsigned short* wsvl = wsvh + nsd;
  unsigned short* wxh  = wsvl + nsd;
  unsigned short* wxl  = wxh + nww;
  unsigned short* wvh  = wxl + nww;
  unsigned short* wvl  = wvh + nww;

  split_weights_kernel<<<dim3((unsigned)(nww / 1024)), 256, 0, stream>>>(
      wxm, wvm, wxh, wxl, wvh, wvl);

  fused_heun_kernel<<<dim3(BATCH / 4), 256, 0, stream>>>(
      xg, vg, fg, lnxg, lnxb, lnvg, lnvb, gw1, gb1, gw2, gb2,
      cu, cv, cw, dtp, wsxh, wsxl, wsvh, wsvl);

  geo_gemm_kernel<<<dim3(DIM / BN, BATCH / BM, 2), 256, 0, stream>>>(
      wsxh, wsxl, wsvh, wsvl, wxh, wxl, wvh, wvl, bxv, bvv, (float*)d_out);
}
